// ModularUnit_73117523247521
// MI455X (gfx1250) — matrix operand order repaired and checked by reading, not compiled, not run on hardware
//
#include <hip/hip_runtime.h>

// ---------------------------------------------------------------------------
// LIF recurrent network, T=1024 steps, N=4096 neurons.
//  Phase 0: convert W_in and spikes to bf16; transpose W_rec to fp32 W_recT.
//  Phase 1: FF[T,N] = spikes @ W_in^T  as a bf16 WMMA GEMM (v_wmma_f32_16x16x32_bf16).
//  Phase 2: 1024 sequential step kernels; recurrent term uses binary-spike
//           row-gather from W_recT (exact fp32, streams from L2).
// ---------------------------------------------------------------------------

#define T_STEPS 1024
#define NND     4096

typedef __attribute__((ext_vector_type(16))) __bf16 v16bf;
typedef __attribute__((ext_vector_type(8)))  __bf16 v8bf;
typedef __attribute__((ext_vector_type(8)))  float  v8f;

// ---- fp32 -> bf16 elementwise -------------------------------------------------
__global__ void cvt_bf16_kernel(const float* __restrict__ in, __bf16* __restrict__ out, int n) {
    int i = blockIdx.x * blockDim.x + threadIdx.x;
    if (i < n) out[i] = (__bf16)in[i];
}

// ---- 4096x4096 fp32 transpose: out[m][n] = in[n][m] --------------------------
__global__ void transpose_f32_kernel(const float* __restrict__ in, float* __restrict__ out) {
    __shared__ float tile[32][33];
    int bx = blockIdx.x * 32;   // column block of `in`
    int by = blockIdx.y * 32;   // row block of `in`
    int tx = threadIdx.x;       // 0..31
    int ty = threadIdx.y;       // 0..7
    for (int r = 0; r < 32; r += 8)
        tile[ty + r][tx] = in[(size_t)(by + ty + r) * NND + (bx + tx)];
    __syncthreads();
    for (int r = 0; r < 32; r += 8)
        out[(size_t)(bx + ty + r) * NND + (by + tx)] = tile[tx][ty + r];
}

// ---- state init ---------------------------------------------------------------
__global__ void lif_init_kernel(float* __restrict__ v, float* __restrict__ cur) {
    int n = blockIdx.x * blockDim.x + threadIdx.x;
    v[n]   = -70.0f;   // V_LEAK
    cur[n] = 0.0f;
}

// ---- FF = S(bf16)[T,K] x W(bf16)[N,K]^T  via WMMA -----------------------------
// Per wave: 2x2 tiles of 16x16 (32x32 output region). Block: 8 waves = 64x128.
// A fragment (16x32, contiguous k order, same as B): lane half h, row m = m0 + (lane&15),
//   v0..7 = K[16h .. 16h+15]                            -> two 16B loads.
// B fragment (32x16, B[k][n] = W[n][k]): lane half h, col n = n0 + (lane&15),
//   v0..7 = W[n][k0+16h .. k0+16h+15]                    -> one 32B load.
struct BfHalves { v8bf lo, hi; };
union BfFrag { v16bf v; BfHalves h; };

__global__ void __launch_bounds__(256)
ff_gemm_bf16_kernel(const __bf16* __restrict__ A,   // [T, K] spikes bf16
                    const __bf16* __restrict__ B,   // [N, K] W_in bf16
                    float* __restrict__ D)          // [T, N] feed-forward currents
{
    const int K    = NND;
    const int lane = threadIdx.x & 31;
    const int w    = threadIdx.x >> 5;     // wave 0..7
    const int l    = lane & 15;
    const int half = lane >> 4;

    const int m0 = blockIdx.y * 64  + (w & 1) * 32;
    const int n0 = blockIdx.x * 128 + (w >> 1) * 32;

    v8f c00 = {}, c01 = {}, c10 = {}, c11 = {};

    const __bf16* arow0 = A + (size_t)(m0 + l)      * K;
    const __bf16* arow1 = A + (size_t)(m0 + 16 + l) * K;
    const __bf16* brow0 = B + (size_t)(n0 + l)      * K;
    const __bf16* brow1 = B + (size_t)(n0 + 16 + l) * K;

    for (int k0 = 0; k0 < K; k0 += 32) {
        BfFrag a0, a1;
        a0.h.lo = *(const v8bf*)(arow0 + k0 +     half * 16);
        a0.h.hi = *(const v8bf*)(arow0 + k0 + 8 + half * 16);
        a1.h.lo = *(const v8bf*)(arow1 + k0 +     half * 16);
        a1.h.hi = *(const v8bf*)(arow1 + k0 + 8 + half * 16);
        v16bf b0 = *(const v16bf*)(brow0 + k0 + half * 16);
        v16bf b1 = *(const v16bf*)(brow1 + k0 + half * 16);

        c00 = __builtin_amdgcn_wmma_f32_16x16x32_bf16(false, a0.v, false, b0, (short)0, c00, false, false);
        c01 = __builtin_amdgcn_wmma_f32_16x16x32_bf16(false, a0.v, false, b1, (short)0, c01, false, false);
        c10 = __builtin_amdgcn_wmma_f32_16x16x32_bf16(false, a1.v, false, b0, (short)0, c10, false, false);
        c11 = __builtin_amdgcn_wmma_f32_16x16x32_bf16(false, a1.v, false, b1, (short)0, c11, false, false);
    }

    // C/D f32 layout: VGPR r -> row (8*half + r), col = lane&15.
    for (int r = 0; r < 8; ++r) {
        const int mrow = half * 8 + r;
        D[(size_t)(m0 + mrow)      * NND + (n0 + l)]      = c00[r];
        D[(size_t)(m0 + mrow)      * NND + (n0 + 16 + l)] = c01[r];
        D[(size_t)(m0 + 16 + mrow) * NND + (n0 + l)]      = c10[r];
        D[(size_t)(m0 + 16 + mrow) * NND + (n0 + 16 + l)] = c11[r];
    }
}

// ---- one LIF timestep ---------------------------------------------------------
// Each block: deterministically compact z_prev (binary) into an LDS spike list,
// then each thread n accumulates sum over spiking m of WrecT[m][n] (coalesced,
// L2-resident) and applies the LIF update. i_t depends on z_{t-1} only, so one
// kernel per step is a sufficient sync point.
__global__ void __launch_bounds__(256)
lif_step_kernel(const float* __restrict__ FFt,     // FF row t            [N]
                const float* __restrict__ WrecT,   // W_rec^T             [N,N]
                const float* __restrict__ zprev,   // z_{t-1} (or null)   [N]
                float* __restrict__ zout,          // z_t  -> d_out row t [N]
                float* __restrict__ vstate,        // membrane voltage    [N]
                float* __restrict__ istate)        // synaptic current    [N]
{
    __shared__ int s_idx[NND];
    __shared__ int s_cnt[257];

    const int tid  = threadIdx.x;
    const int base = tid * (NND / 256);

    // pass 1: per-thread spike counts (ordered, deterministic)
    int lc = 0;
    if (zprev) {
        for (int j = 0; j < NND / 256; ++j)
            if (zprev[base + j] != 0.0f) ++lc;
    }
    s_cnt[tid] = lc;
    __syncthreads();
    if (tid == 0) {
        int run = 0;
        for (int b = 0; b < 256; ++b) { int c = s_cnt[b]; s_cnt[b] = run; run += c; }
        s_cnt[256] = run;
    }
    __syncthreads();
    // pass 2: ordered write of spike indices
    if (zprev) {
        int off = s_cnt[tid];
        for (int j = 0; j < NND / 256; ++j)
            if (zprev[base + j] != 0.0f) s_idx[off++] = base + j;
    }
    __syncthreads();
    const int nspk = s_cnt[256];

    const int n = blockIdx.x * blockDim.x + tid;

    // recurrent current: sum of W_recT rows for spiking neurons (exact fp32)
    float acc = 0.0f;
    int j = 0;
    for (; j + 4 <= nspk; j += 4) {
        int i0 = s_idx[j], i1 = s_idx[j + 1], i2 = s_idx[j + 2], i3 = s_idx[j + 3];
        acc += WrecT[(size_t)i0 * NND + n];
        acc += WrecT[(size_t)i1 * NND + n];
        acc += WrecT[(size_t)i2 * NND + n];
        acc += WrecT[(size_t)i3 * NND + n];
    }
    for (; j < nspk; ++j) acc += WrecT[(size_t)s_idx[j] * NND + n];

    // LIF update (norse defaults)
    const float vv = vstate[n];
    const float ii = istate[n];
    const float v_dec = vv + 5.0e-5f * ((-70.0f - vv) + ii);  // DT*TAU_MEM_INV = 1e-3/20
    const float i_dec = ii * (1.0f - 1.0e-4f);                // DT*TAU_SYN_INV = 1e-3/10
    const float z     = (v_dec - (-55.0f) > 0.0f) ? 1.0f : 0.0f;
    const float v_new = (z != 0.0f) ? -70.0f : v_dec;
    const float i_new = i_dec + FFt[n] + acc;

    zout[n]   = z;
    vstate[n] = v_new;
    istate[n] = i_new;
}

// ---------------------------------------------------------------------------
extern "C" void kernel_launch(void* const* d_in, const int* in_sizes, int n_in,
                              void* d_out, int out_size, void* d_ws, size_t ws_size,
                              hipStream_t stream) {
    const float* spikes = (const float*)d_in[0];   // [T, N] fp32 (0/1)
    const float* W_in   = (const float*)d_in[1];   // [N, N] fp32
    const float* W_rec  = (const float*)d_in[2];   // [N, N] fp32
    float*       zs     = (float*)d_out;           // [T, N] fp32 spikes out

    const size_t NW = (size_t)NND * NND;           // 16.78M
    const size_t NS = (size_t)T_STEPS * NND;       // 4.19M

    // workspace layout (rebuilt every call; no cross-call state)
    char*   ws     = (char*)d_ws;
    __bf16* Wb     = (__bf16*)(ws);                                  // 32 MB
    __bf16* Sb     = (__bf16*)(ws + NW * 2);                         //  8 MB
    float*  FF     = (float*) (ws + NW * 2 + NS * 2);                // 16 MB
    float*  WrecT  = (float*) (ws + NW * 2 + NS * 2 + NS * 4);       // 64 MB
    float*  vstate = (float*) (ws + NW * 2 + NS * 2 + NS * 4 + NW * 4);
    float*  istate = vstate + NND;

    // Phase 0: precision conversion + transpose
    cvt_bf16_kernel<<<(int)((NW + 255) / 256), 256, 0, stream>>>(W_in, Wb, (int)NW);
    cvt_bf16_kernel<<<(int)((NS + 255) / 256), 256, 0, stream>>>(spikes, Sb, (int)NS);
    transpose_f32_kernel<<<dim3(NND / 32, NND / 32), dim3(32, 8), 0, stream>>>(W_rec, WrecT);
    lif_init_kernel<<<NND / 256, 256, 0, stream>>>(vstate, istate);

    // Phase 1: FF = spikes @ W_in^T  (WMMA bf16 GEMM, block tile 64x128)
    ff_gemm_bf16_kernel<<<dim3(NND / 128, T_STEPS / 64), 256, 0, stream>>>(Sb, Wb, FF);

    // Phase 2: sequential LIF recurrence; kernel boundary = grid-wide sync
    for (int t = 0; t < T_STEPS; ++t) {
        const float* zprev = (t == 0) ? nullptr : (zs + (size_t)(t - 1) * NND);
        lif_step_kernel<<<NND / 256, 256, 0, stream>>>(
            FF + (size_t)t * NND, WrecT, zprev, zs + (size_t)t * NND, vstate, istate);
    }
}
